// CKConv_87625922773428
// MI455X (gfx1250) — matrix operand order repaired and checked by reading, not compiled, not run on hardware
//
#include <hip/hip_runtime.h>

typedef __attribute__((ext_vector_type(16))) _Float16 v16h;
typedef __attribute__((ext_vector_type(8)))  float    v8f;

#define OMEGA 30.0f
#define LLEN 2048
#define CH 32
#define PADROWS 64                  // covers max causal overshoot in a quad (63)
#define XT_ROWS (LLEN + PADROWS)    // 2112
#define CHUNK 64                    // lags per wave-chunk (2 waves/SIMD occupancy)
#define NCHUNKMAX 32                // ceil(2048/64)
#define NQUAD 32                    // 32 quads of 4 t-tiles cover L=2048
#define NB 4

// workspace layout (bytes)
#define APACK_ELEMS ((size_t)LLEN * 2 * 32 * 16)       // [d][otile][lane][16] f16
#define APACK_BYTES (APACK_ELEMS * 2)                  // 4 MiB
#define XT_ELEMS    ((size_t)NB * XT_ROWS * CH)
#define XT_BYTES    (XT_ELEMS * 2)
#define PART_OFF    (((APACK_BYTES + XT_BYTES) + 63) & ~(size_t)63)

// ---------------------------------------------------------------------------
// Phase 1a: SIREN kernel generator -> Apack in WMMA A-operand (16x32 f16) layout
// Apack[((d*2 + otile)*32 + lane)*16 + j] = Kr[o][c][d] = kernel[o][c][2047-d]
//   lane<16 : o = otile*16+lane,    j->c: c=j
//   lane>=16: o = otile*16+lane-16, j->c: c=j+16
// ---------------------------------------------------------------------------
__global__ void ck_gen_kernel(const float* __restrict__ w1, const float* __restrict__ b1,
                              const float* __restrict__ w2, const float* __restrict__ b2,
                              const float* __restrict__ w3, const float* __restrict__ b3,
                              _Float16* __restrict__ Apack) {
  __shared__ float h1s[32][16];
  __shared__ float h2s[32][16];
  const int tid = threadIdx.x;          // 0..255
  const int l0  = blockIdx.x * 16;      // 128 blocks cover L=2048

  for (int k = 0; k < 2; ++k) {
    int idx = tid + k * 256;
    int j = idx >> 4, ll = idx & 15;
    float pos = -1.0f + 2.0f * (float)(l0 + ll) * (1.0f / 2047.0f);
    h1s[j][ll] = sinf(OMEGA * (w1[j] * pos + b1[j]));
  }
  __syncthreads();
  for (int k = 0; k < 2; ++k) {
    int idx = tid + k * 256;
    int i = idx >> 4, ll = idx & 15;
    float acc = b2[i];
    #pragma unroll
    for (int j = 0; j < 32; ++j) acc += w2[i * 32 + j] * h1s[j][ll];
    h2s[i][ll] = sinf(OMEGA * acc);
  }
  __syncthreads();
  for (int k = 0; k < 4; ++k) {
    int oc = tid + k * 256;                  // 0..1023
    int o = oc >> 5, c = oc & 31;
    int otile = o >> 4, m = o & 15;
    int lane, j;
    if (c < 16) { lane = m;      j = c;      }
    else        { lane = m + 16; j = c - 16; }
    const float b3v = b3[oc];
    for (int ll = 0; ll < 16; ++ll) {
      float acc = b3v;
      #pragma unroll
      for (int i = 0; i < 32; ++i) acc += w3[oc * 32 + i] * h2s[i][ll];
      int d = 2047 - (l0 + ll);
      Apack[((size_t)(d * 2 + otile) * 32 + lane) * 16 + j] = (_Float16)acc;
    }
  }
}

// ---------------------------------------------------------------------------
// Phase 1b: x -> xT[b][s+64][c] f16, 64 zero rows of causal left-pad
// ---------------------------------------------------------------------------
__global__ void ck_xpack_kernel(const float* __restrict__ x, _Float16* __restrict__ xT) {
  int idx = blockIdx.x * blockDim.x + threadIdx.x;
  if (idx >= NB * XT_ROWS * CH) return;
  int c = idx & 31;
  int r = (idx >> 5) % XT_ROWS;
  int b = idx / (XT_ROWS * CH);
  float v = 0.0f;
  if (r >= PADROWS) v = x[((size_t)(b * CH + c)) * LLEN + (r - PADROWS)];
  xT[idx] = (_Float16)v;
}

// ---------------------------------------------------------------------------
// Phase 2: quad-tile lag-chunked WMMA GEMM. One wave per (b, quad, chunk).
// 4 t-tiles x 2 o-tiles = 8 accumulators; per lag: 2 A + 4 B loads -> 8 WMMAs.
// Pointer-incremental addressing (A += 2KB, B -= one 64B row per lag) and a
// register double-buffer so loads fly behind the matrix pipe.
// Causal masking via the 64-row zero pad of xT (no predication; EXEC all-1s).
// ---------------------------------------------------------------------------
__global__ void ck_conv_kernel(const _Float16* __restrict__ Apack,
                               const _Float16* __restrict__ xT,
                               float* __restrict__ partial) {
  const int lane  = threadIdx.x & 31;
  const int q = blockIdx.x, chunk = blockIdx.y, b = blockIdx.z;
  const int tq0 = q * 64;                  // first output time of the quad
  const int nd  = tq0 + 64;                // lags processed for this quad
  const int d0  = chunk * CHUNK;
  if (d0 >= nd) return;                    // uniform over wave
  const int d1 = (nd < d0 + CHUNK) ? nd : (d0 + CHUNK);

  const int ln   = lane & 15;              // output-time column within a tile
  const int coff = (lane >> 4) * 16;       // channel half for B operand

  // A: [(d*2+otile)*32 + lane]*16 ; stride per lag = 1024 f16 (2 KB)
  const _Float16* ap = Apack + (size_t)d0 * 1024 + (size_t)lane * 16;
  // B tile i at lag d: row (tq0 + 16*i - d + ln + PADROWS); stride per lag = -CH
  const _Float16* bp = xT + (size_t)b * XT_ROWS * CH
                     + (size_t)(tq0 - d0 + ln + PADROWS) * CH + coff;

  v8f acc[8];
  #pragma unroll
  for (int k = 0; k < 8; ++k) acc[k] = (v8f){};

  // pipeline prologue: operands for d0
  v16h a0 = *(const v16h*)(ap);
  v16h a1 = *(const v16h*)(ap + 512);
  v16h bm0 = *(const v16h*)(bp);
  v16h bm1 = *(const v16h*)(bp + 16 * CH);
  v16h bm2 = *(const v16h*)(bp + 32 * CH);
  v16h bm3 = *(const v16h*)(bp + 48 * CH);

  for (int d = d0; d < d1; ++d) {
    ap += 1024;
    bp -= CH;
    __builtin_prefetch(ap + 16 * 1024, 0, 1);   // A stream, ~16 lags (32 KB) ahead
    // next-iteration operands (one-past-end loads land in valid ws memory)
    v16h na0 = *(const v16h*)(ap);
    v16h na1 = *(const v16h*)(ap + 512);
    v16h nb0 = *(const v16h*)(bp);
    v16h nb1 = *(const v16h*)(bp + 16 * CH);
    v16h nb2 = *(const v16h*)(bp + 32 * CH);
    v16h nb3 = *(const v16h*)(bp + 48 * CH);
    // 8 independent WMMA chains on current operands
    acc[0] = __builtin_amdgcn_wmma_f32_16x16x32_f16(false, a0, false, bm0, (short)0, acc[0], false, false);
    acc[1] = __builtin_amdgcn_wmma_f32_16x16x32_f16(false, a1, false, bm0, (short)0, acc[1], false, false);
    acc[2] = __builtin_amdgcn_wmma_f32_16x16x32_f16(false, a0, false, bm1, (short)0, acc[2], false, false);
    acc[3] = __builtin_amdgcn_wmma_f32_16x16x32_f16(false, a1, false, bm1, (short)0, acc[3], false, false);
    acc[4] = __builtin_amdgcn_wmma_f32_16x16x32_f16(false, a0, false, bm2, (short)0, acc[4], false, false);
    acc[5] = __builtin_amdgcn_wmma_f32_16x16x32_f16(false, a1, false, bm2, (short)0, acc[5], false, false);
    acc[6] = __builtin_amdgcn_wmma_f32_16x16x32_f16(false, a0, false, bm3, (short)0, acc[6], false, false);
    acc[7] = __builtin_amdgcn_wmma_f32_16x16x32_f16(false, a1, false, bm3, (short)0, acc[7], false, false);
    // rotate double buffer
    a0 = na0; a1 = na1; bm0 = nb0; bm1 = nb1; bm2 = nb2; bm3 = nb3;
  }

  #pragma unroll
  for (int i = 0; i < 4; ++i) {
    #pragma unroll
    for (int ot = 0; ot < 2; ++ot) {
      float* P = partial +
          ((((size_t)(b * NQUAD + q) * NCHUNKMAX + chunk) * 4 + i) * 2 + ot) * 256;
      v8f a = acc[i * 2 + ot];
      #pragma unroll
      for (int j = 0; j < 8; ++j) P[j * 32 + lane] = a[j];
    }
  }
}

// ---------------------------------------------------------------------------
// Phase 3: reduce lag-chunk partials + bias -> out[b][o][t]
// C/D layout: VGPR j, lanes 0-15: M=j, N=lane ; lanes 16-31: M=j+8, N=lane-16
// ---------------------------------------------------------------------------
__global__ void ck_reduce_kernel(const float* __restrict__ partial,
                                 const float* __restrict__ bias,
                                 float* __restrict__ out) {
  int idx = blockIdx.x * blockDim.x + threadIdx.x;   // 0..262143
  if (idx >= NB * 32 * LLEN) return;
  int t = idx % LLEN;
  int o = (idx / LLEN) & 31;
  int b = idx / (32 * LLEN);
  int q = t >> 6;
  int i = (t >> 4) & 3;
  int ln = t & 15;
  int otile = o >> 4, m = o & 15;
  int j, lane;
  if (m < 8) { j = m; lane = ln; } else { j = m - 8; lane = ln + 16; }
  int nch = (q * 64 + 64 + CHUNK - 1) / CHUNK;        // valid chunks for this quad
  const float* P = partial +
      (((size_t)(b * NQUAD + q) * NCHUNKMAX * 4 + i) * 2 + otile) * 256 +
      (size_t)j * 32 + lane;
  float acc = bias[o];
  for (int ch = 0; ch < nch; ++ch) acc += P[(size_t)ch * (4 * 2 * 256)];
  out[idx] = acc;
}

extern "C" void kernel_launch(void* const* d_in, const int* in_sizes, int n_in,
                              void* d_out, int out_size, void* d_ws, size_t ws_size,
                              hipStream_t stream) {
  const float* x    = (const float*)d_in[0];
  const float* w1   = (const float*)d_in[1];
  const float* b1   = (const float*)d_in[2];
  const float* w2   = (const float*)d_in[3];
  const float* b2   = (const float*)d_in[4];
  const float* w3   = (const float*)d_in[5];
  const float* b3   = (const float*)d_in[6];
  const float* bias = (const float*)d_in[7];
  float* out = (float*)d_out;

  char* ws = (char*)d_ws;
  _Float16* Apack   = (_Float16*)ws;
  _Float16* xT      = (_Float16*)(ws + APACK_BYTES);
  float*    partial = (float*)(ws + PART_OFF);

  ck_gen_kernel<<<128, 256, 0, stream>>>(w1, b1, w2, b2, w3, b3, Apack);
  ck_xpack_kernel<<<(NB * XT_ROWS * CH + 255) / 256, 256, 0, stream>>>(x, xT);
  ck_conv_kernel<<<dim3(NQUAD, NCHUNKMAX, NB), 32, 0, stream>>>(Apack, xT, partial);
  ck_reduce_kernel<<<(NB * 32 * LLEN + 255) / 256, 256, 0, stream>>>(partial, bias, out);
}
